// Attention_77000173683316
// MI455X (gfx1250) — matrix operand order repaired and checked by reading, not compiled, not run on hardware
//
#include <hip/hip_runtime.h>
#include <hip/hip_bf16.h>

// ---------------------------------------------------------------------------
// Problem constants (from reference)
// ---------------------------------------------------------------------------
#define DIMX      2048
#define HEAD_DIM  64
#define N_HEADS   16
#define N_KV      4
#define RANKX     8
#define SEQX      2048
#define QCOLS     (N_HEADS * HEAD_DIM)    // 1024
#define KCOLS     (N_KV * HEAD_DIM)       // 256
#define VCOLS     (N_KV * HEAD_DIM)       // 256
#define QKV_COLS  (QCOLS + KCOLS + VCOLS) // 1536

typedef __bf16 bf16_t;
typedef bf16_t v16bf __attribute__((ext_vector_type(16)));
typedef bf16_t v8bf  __attribute__((ext_vector_type(8)));
typedef float  v8f   __attribute__((ext_vector_type(8)));

// ---------------------------------------------------------------------------
// WMMA fragment loaders (bf16 16x16x32, wave32; A/B in contiguous k order, C/D per CDNA5 ISA 7.12.2)
//
// A matrix 16(M) x 32(K):
//   lane<16 : M=lane,    elems = K k0+0..15
//   lane>=16: M=lane-16, elems = K k0+16..31
// B matrix 32(K) x 16(N), source stored row-major as W[n][k]:
//   lane<16 : N=lane,    elems = K k0+0..15  (contiguous 32B)
//   lane>=16: N=lane-16, elems = K k0+16..31 (contiguous 32B)
// C/D 16x16 f32: VGPR r -> row = r + 8*(lane>=16), col = lane&15
// ---------------------------------------------------------------------------
__device__ __forceinline__ v16bf load_frag_a(const bf16_t* __restrict__ base,
                                             int ld, int row0, int k0, int lane) {
  int m  = lane & 15;
  int kk = k0 + ((lane & 16) ? 16 : 0);
  const bf16_t* p = base + (size_t)(row0 + m) * (size_t)ld + kk;
  v8bf lo = *(const v8bf*)(p);
  v8bf hi = *(const v8bf*)(p + 8);
  v16bf r;
#pragma unroll
  for (int i = 0; i < 8; ++i) { r[i] = lo[i]; r[i + 8] = hi[i]; }
  return r;
}

__device__ __forceinline__ v16bf load_frag_b(const bf16_t* __restrict__ base,
                                             int ld, int n0, int k0, int lane) {
  int n  = lane & 15;
  int kk = k0 + ((lane & 16) ? 16 : 0);
  const bf16_t* p = base + (size_t)(n0 + n) * (size_t)ld + kk;
  return *(const v16bf*)(p);
}

#define WMMA_BF16(A, B, C) \
  __builtin_amdgcn_wmma_f32_16x16x32_bf16(false, (A), false, (B), (short)0, (C), false, false)

// ---------------------------------------------------------------------------
// Prep kernels
// ---------------------------------------------------------------------------
__global__ void convert_x_kernel(const float* __restrict__ x, bf16_t* __restrict__ xb, int n) {
  int i = blockIdx.x * blockDim.x + threadIdx.x;
  if (i < n) xb[i] = (bf16_t)x[i];
}

// Weff rows: [0,1024) Q (scaled by 1/sqrt(64)), [1024,1280) K, [1280,1536) V
// Weff[o,d] = w_base[o%64, d] + 2.0 * sum_r w_b[o, r] * w_a[r, d]
__global__ void build_weff_qkv(const float* __restrict__ wq, const float* __restrict__ wq_a,
                               const float* __restrict__ wq_b,
                               const float* __restrict__ wk, const float* __restrict__ wk_a,
                               const float* __restrict__ wk_b,
                               const float* __restrict__ wv, const float* __restrict__ wv_a,
                               const float* __restrict__ wv_b,
                               bf16_t* __restrict__ weff) {
  int idx = blockIdx.x * blockDim.x + threadIdx.x;
  if (idx >= QKV_COLS * DIMX) return;
  int o = idx >> 11;        // /2048
  int d = idx & (DIMX - 1);
  const float *wbase, *wa, *wb;
  int oo; float post = 1.0f;
  if (o < QCOLS)              { wbase = wq; wa = wq_a; wb = wq_b; oo = o;         post = 0.125f; }
  else if (o < QCOLS + KCOLS) { wbase = wk; wa = wk_a; wb = wk_b; oo = o - QCOLS; }
  else                        { wbase = wv; wa = wv_a; wb = wv_b; oo = o - QCOLS - KCOLS; }
  float acc = wbase[(size_t)(oo & 63) * DIMX + d];
  float lora = 0.f;
#pragma unroll
  for (int r = 0; r < RANKX; ++r) lora += wb[oo * RANKX + r] * wa[(size_t)r * DIMX + d];
  acc = (acc + 2.0f * lora) * post;   // LORA_SCALE = 16/8 = 2
  weff[idx] = (bf16_t)acc;
}

// WoEff[e,f] = wo_share[e,f] + wo[e, f%64]
__global__ void build_woeff(const float* __restrict__ wo, const float* __restrict__ wo_share,
                            bf16_t* __restrict__ woeff) {
  int idx = blockIdx.x * blockDim.x + threadIdx.x;
  if (idx >= DIMX * QCOLS) return;
  int e = idx >> 10;          // /1024
  int f = idx & (QCOLS - 1);
  woeff[idx] = (bf16_t)(wo_share[idx] + wo[(size_t)e * HEAD_DIM + (f & 63)]);
}

// ---------------------------------------------------------------------------
// Fused QKV GEMM + RoPE + scale epilogue. 32x64 tile per wave (2 M x 4 N).
// grid: (24 n-blocks of 64 cols, 64 s-blocks of 32 rows), block: 32 (one wave)
// ---------------------------------------------------------------------------
__global__ void qkv_gemm_rope(const bf16_t* __restrict__ Xb, const bf16_t* __restrict__ Weff,
                              const float* __restrict__ freq,   // (SEQ, 32, 2) cos/sin
                              bf16_t* __restrict__ Q,           // (SEQ, 1024)
                              bf16_t* __restrict__ Kb,          // (SEQ, 256)
                              bf16_t* __restrict__ Vt) {        // (256, SEQ) transposed
  int lane   = threadIdx.x;
  int n_base = blockIdx.x * 64;
  int s_base = blockIdx.y * 32;
  v8f acc[2][4] = {};
  for (int k0 = 0; k0 < DIMX; k0 += 32) {
    v16bf a0 = load_frag_a(Xb, DIMX, s_base +  0, k0, lane);
    v16bf a1 = load_frag_a(Xb, DIMX, s_base + 16, k0, lane);
    __builtin_prefetch(Xb + (size_t)(s_base + (lane & 15)) * DIMX + k0 + 64, 0, 3);
#pragma unroll
    for (int t = 0; t < 4; ++t) {
      v16bf b = load_frag_b(Weff, DIMX, n_base + t * 16, k0, lane);
      acc[0][t] = WMMA_BF16(a0, b, acc[0][t]);
      acc[1][t] = WMMA_BF16(a1, b, acc[1][t]);
    }
  }
  int hi = (lane & 16) ? 8 : 0;
  int nl = lane & 15;
#pragma unroll
  for (int st = 0; st < 2; ++st)
#pragma unroll
    for (int t = 0; t < 4; ++t) {
      int o = n_base + t * 16 + nl;
      bool is_v = (o >= QCOLS + KCOLS);    // uniform per tile
#pragma unroll
      for (int r = 0; r < 8; ++r) {
        int s = s_base + st * 16 + r + hi;
        float v = acc[st][t][r];
        float other = __shfl_xor(v, 1, 32);
        float outv;
        if (!is_v) {
          int fi = (o & 63) >> 1;
          float c  = freq[(size_t)s * 64 + fi * 2 + 0];
          float sn = freq[(size_t)s * 64 + fi * 2 + 1];
          outv = ((o & 1) == 0) ? (v * c - other * sn)    // real part
                                : (other * sn + v * c);   // imag part
        } else {
          outv = v;
        }
        bf16_t bv = (bf16_t)outv;
        if (o < QCOLS)              Q[(size_t)s * QCOLS + o] = bv;
        else if (o < QCOLS + KCOLS) Kb[(size_t)s * KCOLS + (o - QCOLS)] = bv;
        else                        Vt[(size_t)(o - QCOLS - KCOLS) * SEQX + s] = bv;
      }
    }
}

// ---------------------------------------------------------------------------
// Flash attention, one wave32 per (q-head, 16-query tile). Causal, key tile 32.
// grid: (16 heads, 128 s-tiles), block: 32
// ---------------------------------------------------------------------------
__global__ void attention_kernel(const bf16_t* __restrict__ Q, const bf16_t* __restrict__ Kb,
                                 const bf16_t* __restrict__ Vt, bf16_t* __restrict__ AttnOut) {
  __shared__ __align__(32) float  s_lds[16 * 32];
  __shared__ __align__(32) bf16_t p_lds[16 * 32];
  __shared__ float m_lds[16], l_lds[16], corr_lds[16];

  int lane   = threadIdx.x;
  int h      = blockIdx.x;
  int s_base = blockIdx.y * 16;
  int kvh    = h >> 2;                // GQA group of 4
  int hi     = (lane & 16) ? 8 : 0;
  int nl     = lane & 15;

  // Q fragments for this tile (scale already folded into Weff_q)
  v16bf qa0 = load_frag_a(Q, QCOLS, s_base, h * HEAD_DIM + 0,  lane);
  v16bf qa1 = load_frag_a(Q, QCOLS, s_base, h * HEAD_DIM + 32, lane);

  if (lane < 16) { m_lds[lane] = -3.0e38f; l_lds[lane] = 0.f; }
  __syncthreads();

  v8f oacc[4] = {};
  int kend = s_base + 16;
  for (int kb = 0; kb < kend; kb += 32) {
    // ---- scores: 16x32 = 2 C-tiles, K-dim 64 = 2 chained WMMAs each
    v8f sc[2] = {};
#pragma unroll
    for (int nt = 0; nt < 2; ++nt) {
      v16bf b0 = load_frag_b(Kb, KCOLS, kb + nt * 16, kvh * HEAD_DIM + 0,  lane);
      v16bf b1 = load_frag_b(Kb, KCOLS, kb + nt * 16, kvh * HEAD_DIM + 32, lane);
      sc[nt] = WMMA_BF16(qa0, b0, sc[nt]);
      sc[nt] = WMMA_BF16(qa1, b1, sc[nt]);
    }
    // ---- causal mask + stage scores to LDS (C-layout -> row-major)
#pragma unroll
    for (int nt = 0; nt < 2; ++nt)
#pragma unroll
      for (int r = 0; r < 8; ++r) {
        int key = kb + nt * 16 + nl;
        int s   = s_base + r + hi;
        s_lds[(r + hi) * 32 + nt * 16 + nl] = (key <= s) ? sc[nt][r] : -3.0e38f;
      }
    __syncthreads();

    // ---- online softmax, one row per lane (lanes 0..15)
    if (lane < 16) {
      float mold = m_lds[lane];
      float mx = mold;
#pragma unroll
      for (int j = 0; j < 32; ++j) mx = fmaxf(mx, s_lds[lane * 32 + j]);
      float corr = __expf(mold - mx);
      float sum = 0.f;
#pragma unroll
      for (int j = 0; j < 32; ++j) {
        float p = __expf(s_lds[lane * 32 + j] - mx);
        sum += p;
        p_lds[lane * 32 + j] = (bf16_t)p;
      }
      m_lds[lane]    = mx;
      l_lds[lane]    = l_lds[lane] * corr + sum;
      corr_lds[lane] = corr;
    }
    __syncthreads();

    // ---- rescale running output by correction factor
#pragma unroll
    for (int r = 0; r < 8; ++r) {
      float c = corr_lds[r + hi];
#pragma unroll
      for (int t = 0; t < 4; ++t) oacc[t][r] *= c;
    }

    // ---- P @ V : A = P (16x32) from LDS, B = Vt rows (contiguous keys)
    v16bf pa = load_frag_a(p_lds, 32, 0, 0, lane);
#pragma unroll
    for (int t = 0; t < 4; ++t) {
      v16bf vb = load_frag_b(Vt, SEQX, kvh * HEAD_DIM + t * 16, kb, lane);
      oacc[t] = WMMA_BF16(pa, vb, oacc[t]);
    }
    __syncthreads();   // s_lds / p_lds reused next iteration
  }

  // ---- normalize and store (row-major bf16 for the output GEMM)
#pragma unroll
  for (int r = 0; r < 8; ++r) {
    float inv = 1.0f / l_lds[r + hi];
    int s = s_base + r + hi;
#pragma unroll
    for (int t = 0; t < 4; ++t) {
      int f = h * HEAD_DIM + t * 16 + nl;
      AttnOut[(size_t)s * QCOLS + f] = (bf16_t)(oacc[t][r] * inv);
    }
  }
}

// ---------------------------------------------------------------------------
// Output GEMM: out[s,e] = sum_f AttnOut[s,f] * WoEff[e,f]  (fp32 result)
// 32x64 tile per wave. grid: (32 e-blocks of 64, 64 s-blocks of 32), block: 32
// ---------------------------------------------------------------------------
__global__ void out_gemm(const bf16_t* __restrict__ AttnOut, const bf16_t* __restrict__ WoEff,
                         float* __restrict__ out) {
  int lane   = threadIdx.x;
  int e_base = blockIdx.x * 64;
  int s_base = blockIdx.y * 32;
  v8f acc[2][4] = {};
  for (int k0 = 0; k0 < QCOLS; k0 += 32) {
    v16bf a0 = load_frag_a(AttnOut, QCOLS, s_base +  0, k0, lane);
    v16bf a1 = load_frag_a(AttnOut, QCOLS, s_base + 16, k0, lane);
    __builtin_prefetch(WoEff + (size_t)(e_base + (lane & 15)) * QCOLS + k0 + 64, 0, 3);
#pragma unroll
    for (int t = 0; t < 4; ++t) {
      v16bf b = load_frag_b(WoEff, QCOLS, e_base + t * 16, k0, lane);
      acc[0][t] = WMMA_BF16(a0, b, acc[0][t]);
      acc[1][t] = WMMA_BF16(a1, b, acc[1][t]);
    }
  }
  int hi = (lane & 16) ? 8 : 0;
  int nl = lane & 15;
#pragma unroll
  for (int st = 0; st < 2; ++st)
#pragma unroll
    for (int t = 0; t < 4; ++t)
#pragma unroll
      for (int r = 0; r < 8; ++r) {
        int s = s_base + st * 16 + r + hi;
        int e = e_base + t * 16 + nl;
        out[(size_t)s * DIMX + e] = acc[st][t][r];
      }
}

// ---------------------------------------------------------------------------
// Launch
// ---------------------------------------------------------------------------
extern "C" void kernel_launch(void* const* d_in, const int* in_sizes, int n_in,
                              void* d_out, int out_size, void* d_ws, size_t ws_size,
                              hipStream_t stream) {
  const float* x        = (const float*)d_in[0];
  const float* freq_cis = (const float*)d_in[1];
  const float* wq       = (const float*)d_in[2];
  const float* wq_a     = (const float*)d_in[3];
  const float* wq_b     = (const float*)d_in[4];
  const float* wk       = (const float*)d_in[5];
  const float* wk_a     = (const float*)d_in[6];
  const float* wk_b     = (const float*)d_in[7];
  const float* wv       = (const float*)d_in[8];
  const float* wv_a     = (const float*)d_in[9];
  const float* wv_b     = (const float*)d_in[10];
  const float* wo       = (const float*)d_in[11];
  const float* wo_share = (const float*)d_in[12];
  float* out = (float*)d_out;

  // Workspace layout (bytes, all 256B-aligned)
  char* w = (char*)d_ws;
  bf16_t* Xb      = (bf16_t*)(w + 0);            // 2048*2048*2 = 8388608
  bf16_t* Weff    = (bf16_t*)(w + 8388608);      // 1536*2048*2 = 6291456
  bf16_t* Qbuf    = (bf16_t*)(w + 14680064);     // 2048*1024*2 = 4194304
  bf16_t* Kbuf    = (bf16_t*)(w + 18874368);     // 2048*256*2  = 1048576
  bf16_t* Vt      = (bf16_t*)(w + 19922944);     // 256*2048*2  = 1048576
  bf16_t* WoEff   = (bf16_t*)(w + 20971520);     // 2048*1024*2 = 4194304
  bf16_t* AttnOut = (bf16_t*)(w + 25165824);     // 2048*1024*2 = 4194304

  // 1) x -> bf16
  {
    int n = SEQX * DIMX;
    convert_x_kernel<<<(n + 255) / 256, 256, 0, stream>>>(x, Xb, n);
  }
  // 2) effective weights (LoRA + base folded, q-scale folded)
  {
    int n = QKV_COLS * DIMX;
    build_weff_qkv<<<(n + 255) / 256, 256, 0, stream>>>(wq, wq_a, wq_b, wk, wk_a, wk_b,
                                                        wv, wv_a, wv_b, Weff);
    int m = DIMX * QCOLS;
    build_woeff<<<(m + 255) / 256, 256, 0, stream>>>(wo, wo_share, WoEff);
  }
  // 3) fused QKV projection + RoPE (32x64 tiles)
  qkv_gemm_rope<<<dim3(QKV_COLS / 64, SEQX / 32), 32, 0, stream>>>(Xb, Weff, freq_cis,
                                                                   Qbuf, Kbuf, Vt);
  // 4) flash attention
  attention_kernel<<<dim3(N_HEADS, SEQX / 16), 32, 0, stream>>>(Qbuf, Kbuf, Vt, AttnOut);
  // 5) output projection (wo folded into wo_share, 32x64 tiles)
  out_gemm<<<dim3(DIMX / 64, SEQX / 32), 32, 0, stream>>>(AttnOut, WoEff, out);
}
